// GNNEncoder_23579370455553
// MI455X (gfx1250) — matrix operand order repaired and checked by reading, not compiled, not run on hardware
//
#include <hip/hip_runtime.h>

// GNN encoder: 3 × (Linear[WMMA f16→f32] → mean-aggregate over edges+self →
// ReLU → LayerNorm). Scatter-bound; working set is L2-resident on MI455X.

#define N_NODES 50000
#define IN_DIM  64
#define HID     128
#define LN_EPS  1e-5f

typedef _Float16 half16 __attribute__((ext_vector_type(16)));
typedef _Float16 half8  __attribute__((ext_vector_type(8)));
typedef float    f32x8  __attribute__((ext_vector_type(8)));
typedef float    f32x4  __attribute__((ext_vector_type(4)));

// ---------------- small utility kernels ----------------

__global__ void k_zero(float* __restrict__ p, int n) {
    int i = blockIdx.x * blockDim.x + threadIdx.x;
    if (i < n) p[i] = 0.0f;
}

// degree count over edge destinations (self-loop added in k_inv)
__global__ void k_count(const int* __restrict__ dst, float* __restrict__ cnt, int E) {
    int e = blockIdx.x * blockDim.x + threadIdx.x;
    if (e < E) unsafeAtomicAdd(&cnt[dst[e]], 1.0f);
}

__global__ void k_inv(const float* __restrict__ cnt, float* __restrict__ inv, int n) {
    int i = blockIdx.x * blockDim.x + threadIdx.x;
    if (i < n) inv[i] = 1.0f / (cnt[i] + 1.0f);   // +1 = self loop
}

// W [Kdim x HID] f32 (row-major) -> Wt [HID x Kdim] f16 (transposed), so the
// WMMA B-fragment reads K values as two contiguous runs of 8 per lane.
__global__ void k_cvt_w(const float* __restrict__ W, _Float16* __restrict__ Wt, int Kdim) {
    int t = blockIdx.x * blockDim.x + threadIdx.x;
    if (t < Kdim * HID) {
        int k = t / HID, n = t % HID;
        Wt[(size_t)n * Kdim + k] = (_Float16)W[t];
    }
}

// ---------------- WMMA GEMM:  z = A @ W + b ;  s = z (self-loop init) ----------------
// One wave computes a 16-row x 128-col strip; weights staged in LDS per block.
// Kdim in {64,128}. 50000 rows = 3125 exact 16-row tiles.

__global__ __launch_bounds__(256)
void k_gemm(const float* __restrict__ A, int Kdim,
            const _Float16* __restrict__ Wt, const float* __restrict__ bias,
            float* __restrict__ z, float* __restrict__ s, int ntiles) {
    __shared__ _Float16 WtLds[HID * HID];        // up to 32 KB (of 320 KB/WGP)

    // Cooperative weight stage (all waves participate, then barrier).
    const int nchunks = (Kdim * HID) / 8;        // half8 chunks
    for (int i = threadIdx.x; i < nchunks; i += blockDim.x)
        ((half8*)WtLds)[i] = ((const half8*)Wt)[i];
    __syncthreads();

    const int wave = (blockIdx.x * blockDim.x + threadIdx.x) >> 5;
    if (wave >= ntiles) return;                  // wave-uniform: EXEC all-ones for WMMA
    const int lane = threadIdx.x & 31;
    const int hs   = lane >> 4;                  // lane-half select
    const int ln   = lane & 15;
    const int row0 = wave * 16;
    const int m    = row0 + ln;                  // A row for this lane

    // C/D layout: VGPR r, lanes 0-15 -> D[M=r][N=ln]; lanes 16-31 -> D[M=8+r][N=ln].
    // Bias depends only on N -> splat per accumulator.
    f32x8 acc[8];
#pragma unroll
    for (int t = 0; t < 8; ++t) {
        float bv = bias[t * 16 + ln];
#pragma unroll
        for (int r = 0; r < 8; ++r) acc[t][r] = bv;
    }

    for (int kc = 0; kc < Kdim; kc += 32) {
        // A fragment (16x32 f16): lanes 0-15 hold K = kc+{0..7, 16..23},
        // lanes 16-31 hold K = kc+{8..15, 24..31}, row m = row0 + ln.
        const int ka = kc + hs * 8;
        const float* ap = A + (size_t)m * Kdim + ka;
        f32x4 a0 = *(const f32x4*)(ap);
        f32x4 a1 = *(const f32x4*)(ap + 4);
        f32x4 a2 = *(const f32x4*)(ap + 16);
        f32x4 a3 = *(const f32x4*)(ap + 20);
        half16 afrag;
#pragma unroll
        for (int i = 0; i < 4; ++i) {
            afrag[i]      = (_Float16)a0[i];
            afrag[4 + i]  = (_Float16)a1[i];
            afrag[8 + i]  = (_Float16)a2[i];
            afrag[12 + i] = (_Float16)a3[i];
        }

        // B fragment (32x16 f16) from LDS: lanes 0-15 hold K = kc+{0..7, 16..23},
        // lanes 16-31 hold K = kc+{8..15, 24..31}, column n = t*16 + ln.
        const int kb = kc + hs * 8;
#pragma unroll
        for (int t = 0; t < 8; ++t) {
            const _Float16* bp = WtLds + (t * 16 + ln) * Kdim + kb;
            half8 b0 = *(const half8*)bp;
            half8 b1 = *(const half8*)(bp + 16);
            half16 bfrag;
#pragma unroll
            for (int i = 0; i < 8; ++i) { bfrag[i] = b0[i]; bfrag[8 + i] = b1[i]; }
            acc[t] = __builtin_amdgcn_wmma_f32_16x16x32_f16(
                false, afrag, false, bfrag, (short)0, acc[t], false, false);
        }
    }

    // Dual-store: z (gather source) and s (accumulator pre-seeded with self-loop).
    // Half-wave writes 64B contiguous per (t,r): full 128B payload per instruction.
#pragma unroll
    for (int t = 0; t < 8; ++t) {
        const int col = t * 16 + ln;
#pragma unroll
        for (int r = 0; r < 8; ++r) {
            const int row = row0 + hs * 8 + r;
            const size_t o = (size_t)row * HID + col;
            float v = acc[t][r];
            z[o] = v;
            s[o] = v;
        }
    }
}

// ---------------- edge scatter:  s[dst] += z[src]  ----------------
// One wave per 32 edges: coalesced index loads, v_readlane broadcast, and a
// software-pipelined b128 gather overlapping 4 native f32 atomics per lane.

__global__ __launch_bounds__(256)
void k_edge(const int* __restrict__ src, const int* __restrict__ dst,
            const float* __restrict__ z, float* __restrict__ s, int E) {
    const int wave = (blockIdx.x * blockDim.x + threadIdx.x) >> 5;
    const int lane = threadIdx.x & 31;
    const int e0 = wave << 5;
    if (e0 >= E) return;                          // wave-uniform
    const int nedge = (E - e0 < 32) ? (E - e0) : 32;
    const int eL = e0 + ((lane < nedge) ? lane : 0);
    const int sv = src[eL];                       // one edge's indices per lane
    const int dv = dst[eL];

    int sn = __shfl(sv, 0, 32);
    int dn = __shfl(dv, 0, 32);
    f32x4 v = *(const f32x4*)(z + (size_t)sn * HID + (lane << 2));
    for (int i = 0; i < nedge; ++i) {
        const f32x4 vc = v;
        const int   dc = dn;
        if (i + 1 < nedge) {                      // prefetch next gather
            sn = __shfl(sv, i + 1, 32);
            dn = __shfl(dv, i + 1, 32);
            v = *(const f32x4*)(z + (size_t)sn * HID + (lane << 2));
        }
        float* p = s + (size_t)dc * HID + (lane << 2);
        unsafeAtomicAdd(p + 0, vc[0]);
        unsafeAtomicAdd(p + 1, vc[1]);
        unsafeAtomicAdd(p + 2, vc[2]);
        unsafeAtomicAdd(p + 3, vc[3]);
    }
}

// ---------------- fused mean-scale + ReLU + LayerNorm ----------------
// One wave per node (128 feats = 4/lane), wave32 butterfly reduction.

__global__ __launch_bounds__(256)
void k_post(const float* __restrict__ s, const float* __restrict__ inv,
            const float* __restrict__ g, const float* __restrict__ be,
            float* __restrict__ out, int n) {
    const int node = (blockIdx.x * blockDim.x + threadIdx.x) >> 5;
    if (node >= n) return;
    const int lane = threadIdx.x & 31;
    const size_t base = (size_t)node * HID + lane * 4;

    f32x4 v = *(const f32x4*)(s + base);
    const float ic = inv[node];
#pragma unroll
    for (int j = 0; j < 4; ++j) {
        float t = v[j] * ic;
        v[j] = t > 0.0f ? t : 0.0f;             // mean aggregate + ReLU
    }
    float sum = v[0] + v[1] + v[2] + v[3];
    float sq  = v[0] * v[0] + v[1] * v[1] + v[2] * v[2] + v[3] * v[3];
#pragma unroll
    for (int off = 16; off > 0; off >>= 1) {
        sum += __shfl_xor(sum, off, 32);
        sq  += __shfl_xor(sq,  off, 32);
    }
    const float mu  = sum * (1.0f / HID);
    const float var = sq * (1.0f / HID) - mu * mu;
    const float rs  = rsqrtf(var + LN_EPS);
    f32x4 o;
#pragma unroll
    for (int j = 0; j < 4; ++j) {
        const int c = lane * 4 + j;
        o[j] = g[c] * (v[j] - mu) * rs + be[c];
    }
    *(f32x4*)(out + base) = o;
}

// ---------------- launcher ----------------

extern "C" void kernel_launch(void* const* d_in, const int* in_sizes, int n_in,
                              void* d_out, int out_size, void* d_ws, size_t ws_size,
                              hipStream_t stream) {
    (void)n_in; (void)out_size; (void)ws_size;
    const float* x  = (const float*)d_in[0];
    const int*   ei = (const int*)d_in[1];          // harness: integer -> const int*
    const int E = in_sizes[1] / 2;                  // edge_index is [2, E]
    const int* srcp = ei;
    const int* dstp = ei + E;
    const float* W[3]  = {(const float*)d_in[2], (const float*)d_in[4], (const float*)d_in[6]};
    const float* B[3]  = {(const float*)d_in[3], (const float*)d_in[5], (const float*)d_in[7]};
    const float* G[3]  = {(const float*)d_in[8], (const float*)d_in[10], (const float*)d_in[12]};
    const float* Be[3] = {(const float*)d_in[9], (const float*)d_in[11], (const float*)d_in[13]};

    // workspace layout (all re-written deterministically every call)
    const size_t NH = (size_t)N_NODES * HID;
    float* z    = (float*)d_ws;          // [N, HID] linear output (gather source)
    float* sbuf = z + NH;                // [N, HID] scatter accumulator
    float* h    = sbuf + NH;             // [N, HID] layer activation
    float* cnt  = h + NH;                // [N] degree
    float* inv  = cnt + N_NODES;         // [N] 1/(deg+1)
    _Float16* Wt = (_Float16*)(inv + N_NODES);  // [HID, Kdim] f16 transposed weights

    // degree / inv_cnt (layer-invariant)
    k_zero<<<(N_NODES + 255) / 256, 256, 0, stream>>>(cnt, N_NODES);
    k_count<<<(E + 255) / 256, 256, 0, stream>>>(dstp, cnt, E);
    k_inv<<<(N_NODES + 255) / 256, 256, 0, stream>>>(cnt, inv, N_NODES);

    const int ntiles = N_NODES / 16;     // 3125 exact
    const int WPB = 256 / 32;            // waves per block
    const int ewaves = (E + 31) / 32;    // 32 edges per wave
    for (int l = 0; l < 3; ++l) {
        const int Kdim = (l == 0) ? IN_DIM : HID;
        k_cvt_w<<<(Kdim * HID + 255) / 256, 256, 0, stream>>>(W[l], Wt, Kdim);
        const float* Ain = (l == 0) ? x : h;
        k_gemm<<<(ntiles + WPB - 1) / WPB, 256, 0, stream>>>(Ain, Kdim, Wt, B[l], z, sbuf, ntiles);
        k_edge<<<(ewaves + WPB - 1) / WPB, 256, 0, stream>>>(srcp, dstp, z, sbuf, E);
        float* outp = (l == 2) ? (float*)d_out : h;
        k_post<<<(N_NODES + WPB - 1) / WPB, 256, 0, stream>>>(sbuf, inv, G[l], Be[l], outp, N_NODES);
    }
}
